// EfficientAdaptiveGraphAttentionModule_30081950941207
// MI455X (gfx1250) — matrix operand order repaired and checked by reading, not compiled, not run on hardware
//
#include <hip/hip_runtime.h>

// ---------------------------------------------------------------------------
// MI455X (gfx1250, wave32) implementation.
//
// Key algebra:
//  * attn_reg_loss = REG_W * mean(|softmax(...)|) = REG_W * (1/N)  (rows of a
//    softmax sum to 1, outputs are >= 0)  ->  constant; adj_u/adj_v unused.
//  * low-rank pairs fused offline:  Weff  = w_qkv_high @ w_qkv_low   [768,256]
//                                   Wout  = w_out_high @ w_out_low   [256,256]
// Pipeline (all WMMA bf16 x bf16 -> f32):
//  1) prep:   Weff/beff, Wout/bout                           (tiny, VALU)
//  2) qkv:    [8192,256] x [256,768] -> qf / kfT / vT (bf16, feature map fused)
//  3) zsum:   z[bh,n] = 1/(sum_d kf + eps)
//  4) kv:     per (b,h): kfT[64,2048] x vT^T -> kvT[64,64]
//             (K-loop staged through LDS with async-to-LDS double buffering)
//  5) attn:   per (b,h): qf[2048,64] x kv -> lin (scaled by z, heads merged)
//  6) proj:   [8192,256] x [256,256] + bias -> out (f32) ; writes loss scalar
// ---------------------------------------------------------------------------

typedef __bf16 bf16;
typedef __attribute__((ext_vector_type(16))) __bf16 v16bf;
typedef __attribute__((ext_vector_type(8)))  __bf16 v8bf;
typedef __attribute__((ext_vector_type(8)))  float  v8f;

#define NSEQ   2048
#define HID    256
#define NHEAD  4
#define HDIM   64
#define NBATCH 4
#define NROWS  (NBATCH * NSEQ)          // 8192
#define NBH    (NBATCH * NHEAD)         // 16

__device__ __forceinline__ float elu1(float x) {
    return x > 0.0f ? x + 1.0f : __expf(x);
}

// ---- WMMA fragment loaders (wave32, 16x16x32 bf16) -------------------------
// A: row-major [M][K].  Lane l holds row M=l%16; halves 0..7 = K base+(l/16)*8,
// halves 8..15 = same +16 (ISA 7.12.2, 16-bit A 16x32).
__device__ __forceinline__ v16bf ldA(const bf16* base, int ld, int row0,
                                     int col0, int lane) {
    const bf16* p = base + (row0 + (lane & 15)) * ld + col0 + ((lane >> 4) << 3);
    v8bf lo = *(const v8bf*)p;
    v8bf hi = *(const v8bf*)(p + 16);
    v16bf a;
#pragma unroll
    for (int i = 0; i < 8; ++i) { a[i] = lo[i]; a[i + 8] = hi[i]; }
    return a;
}

// A from f32 source (convert to bf16 in registers).
__device__ __forceinline__ v16bf ldA_f32(const float* base, int ld, int row0,
                                         int col0, int lane) {
    const float* p = base + (row0 + (lane & 15)) * ld + col0 + ((lane >> 4) << 3);
    v16bf a;
#pragma unroll
    for (int i = 0; i < 8; ++i) { a[i] = (bf16)p[i]; a[i + 8] = (bf16)p[i + 16]; }
    return a;
}

// B: stored as Bt[N][K] row-major. Lane l holds column N=l%16 of B; halves
// 0..7 = K col0+(l/16)*8, halves 8..15 = same +16 (matches A order).
__device__ __forceinline__ v16bf ldB(const bf16* baseT, int ld, int row0,
                                     int col0, int lane) {
    const bf16* p = baseT + (row0 + (lane & 15)) * ld + col0 + ((lane >> 4) << 3);
    v8bf lo = *(const v8bf*)p;
    v8bf hi = *(const v8bf*)(p + 16);
    v16bf b;
#pragma unroll
    for (int i = 0; i < 8; ++i) { b[i] = lo[i]; b[i + 8] = hi[i]; }
    return b;
}

__device__ __forceinline__ v8f wmma_bf16(v16bf a, v16bf b, v8f c) {
    return __builtin_amdgcn_wmma_f32_16x16x32_bf16(false, a, false, b,
                                                   (short)0, c, false, false);
}

// ---- async global -> LDS staging (CDNA5) -----------------------------------
__device__ __forceinline__ unsigned lds_off(const void* p) {
    // generic pointer to __shared__: low 32 bits are the LDS byte address
    return (unsigned)(unsigned long long)p;
}

__device__ __forceinline__ void async_b128(unsigned lds, const void* g) {
    asm volatile("global_load_async_to_lds_b128 %0, %1, off"
                 :: "v"(lds), "v"((unsigned long long)g) : "memory");
}

// Stage a 16-row x 32-col bf16 tile (16 x 64B) from row-major global (ld elems)
// into LDS laid out row-major [16][32].  2 async instructions per wave.
__device__ __forceinline__ void stage16x32(unsigned lds, const bf16* g, int ld,
                                           int lane) {
#pragma unroll
    for (int j = 0; j < 2; ++j) {
        int s = lane + 32 * j;        // 64 segments of 16B
        int row = s >> 2, seg = s & 3;
        async_b128(lds + row * 64 + seg * 16, g + row * ld + seg * 8);
    }
}

// ---------------------------------------------------------------------------
// 1) Fuse the low-rank factor pairs (tiny).
// ---------------------------------------------------------------------------
__global__ void prep_qkv_w(const float* __restrict__ wlo, const float* __restrict__ blo,
                           const float* __restrict__ whi, const float* __restrict__ bhi,
                           bf16* __restrict__ Weff, float* __restrict__ beff) {
    int j = blockIdx.x, i = threadIdx.x;      // [768] x [256]
    float s = 0.0f;
#pragma unroll
    for (int r = 0; r < 24; ++r) s += whi[j * 24 + r] * wlo[r * HID + i];
    Weff[j * HID + i] = (bf16)s;
    if (i == 0) {
        float b = bhi[j];
#pragma unroll
        for (int r = 0; r < 24; ++r) b += whi[j * 24 + r] * blo[r];
        beff[j] = b;
    }
}

__global__ void prep_out_w(const float* __restrict__ wlo, const float* __restrict__ blo,
                           const float* __restrict__ whi, const float* __restrict__ bhi,
                           bf16* __restrict__ Wout, float* __restrict__ bout) {
    int j = blockIdx.x, i = threadIdx.x;      // [256] x [256]
    float s = 0.0f;
#pragma unroll
    for (int r = 0; r < 8; ++r) s += whi[j * 8 + r] * wlo[r * HID + i];
    Wout[j * HID + i] = (bf16)s;
    if (i == 0) {
        float b = bhi[j];
#pragma unroll
        for (int r = 0; r < 8; ++r) b += whi[j * 8 + r] * blo[r];
        bout[j] = b;
    }
}

// ---------------------------------------------------------------------------
// 2) qkv = x @ Weff^T + beff, feature maps fused, layout-transforming epilogue.
//    One wave per 16x16 tile of [8192, 768]; K = 256 -> 8 WMMAs.
// ---------------------------------------------------------------------------
__global__ void qkv_kernel(const float* __restrict__ x, const bf16* __restrict__ Weff,
                           const float* __restrict__ beff,
                           bf16* __restrict__ qf,   // [bh][n][d]
                           bf16* __restrict__ kfT,  // [bh][d][n]
                           bf16* __restrict__ vT) { // [bh][d][n]
    const int lane = threadIdx.x & 31, w = threadIdx.x >> 5;
    const int tile = blockIdx.x * 8 + w;          // 512 * 48 tiles
    const int m0 = tile / 48, j0 = (tile % 48) << 4;
    const int row0 = m0 << 4;
    v8f c = {};
#pragma unroll
    for (int kk = 0; kk < HID; kk += 32) {
        __builtin_prefetch(x + (row0 + (lane & 15)) * HID + kk + 64, 0, 3);
        v16bf a = ldA_f32(x, HID, row0, kk, lane);
        v16bf b = ldB(Weff, HID, j0, kk, lane);
        c = wmma_bf16(a, b, c);
    }
    const int col = lane & 15;
    const float bias = beff[j0 + col];
    const int region = j0 >> 8;                   // 0=q 1=k 2=v
    const int jq = j0 & 255, h = jq >> 6, d0 = jq & 63;
    const int b_ = row0 >> 11, nb = row0 & (NSEQ - 1);
    const int bh = (b_ << 2) | h;
    if (region == 0) {
        // qf row-major [n][d]: lane column d fixed, rows vary with r.
#pragma unroll
        for (int r = 0; r < 8; ++r) {
            int n = nb + r + ((lane >> 4) << 3);
            qf[(bh * NSEQ + n) * HDIM + d0 + col] = (bf16)elu1(c[r] + bias);
        }
    } else {
        // kfT/vT [d][n]: lane row d fixed, n contiguous with r -> packed b128.
        bf16* dst = (region == 1) ? kfT : vT;
        const int d = d0 + col;
        v8bf pk;
#pragma unroll
        for (int r = 0; r < 8; ++r) {
            float v = c[r] + bias;
            pk[r] = (bf16)((region == 1) ? elu1(v) : v);
        }
        *(v8bf*)(dst + (bh * HDIM + d) * NSEQ + nb + ((lane >> 4) << 3)) = pk;
    }
}

// ---------------------------------------------------------------------------
// 3) z[bh,n] = 1 / (sum_d kf[n,d] + eps)
// ---------------------------------------------------------------------------
__global__ void zsum_kernel(const bf16* __restrict__ kfT, float* __restrict__ z) {
    int i = blockIdx.x * blockDim.x + threadIdx.x;   // NBH * NSEQ
    const bf16* p = kfT + (i >> 11) * (HDIM * NSEQ) + (i & (NSEQ - 1));
    float s = 0.0f;
#pragma unroll
    for (int d = 0; d < HDIM; ++d) s += (float)p[d * NSEQ];
    z[i] = 1.0f / (s + 1e-8f);
}

// ---------------------------------------------------------------------------
// 4) kv[bh] = kf^T @ v : per-wave 16x16 tile, K = N = 2048 (64 WMMAs),
//    double-buffered through LDS with async-to-LDS (ASYNCcnt pipeline).
// ---------------------------------------------------------------------------
__global__ void kv_kernel(const bf16* __restrict__ kfT, const bf16* __restrict__ vT,
                          bf16* __restrict__ kvT) {  // [bh][e][d]
    __shared__ __attribute__((aligned(32))) bf16 smem[8 * 2048]; // 4KB/wave
    const int lane = threadIdx.x & 31, w = threadIdx.x >> 5;
    const int tile = blockIdx.x * 8 + w;             // 16 bh * 16 tiles
    const int bh = tile >> 4, t = tile & 15;
    const int d0 = (t >> 2) << 4, e0 = (t & 3) << 4;
    const bf16* Ag = kfT + (bh * HDIM + d0) * NSEQ;  // rows d, ld NSEQ
    const bf16* Bg = vT  + (bh * HDIM + e0) * NSEQ;  // rows e, ld NSEQ
    bf16* lb = smem + w * 2048;                      // A0 A1 B0 B1 (512 elems ea)
    const unsigned aA0 = lds_off(lb),        aA1 = lds_off(lb + 512);
    const unsigned aB0 = lds_off(lb + 1024), aB1 = lds_off(lb + 1536);

    stage16x32(aA0, Ag, NSEQ, lane);
    stage16x32(aB0, Bg, NSEQ, lane);
    v8f c = {};
    for (int k = 0; k < NSEQ / 32; ++k) {
        const int cur = k & 1;
        if (k + 1 < NSEQ / 32) {
            // prefetch next K-chunk while this one is consumed
            stage16x32(cur ? aA0 : aA1, Ag + (k + 1) * 32, NSEQ, lane);
            stage16x32(cur ? aB0 : aB1, Bg + (k + 1) * 32, NSEQ, lane);
            asm volatile("s_wait_asynccnt 0x4" ::: "memory"); // chunk k landed
        } else {
            asm volatile("s_wait_asynccnt 0x0" ::: "memory");
        }
        v16bf a = ldA(lb + cur * 512, 32, 0, 0, lane);         // ds reads
        v16bf b = ldB(lb + 1024 + cur * 512, 32, 0, 0, lane);
        c = wmma_bf16(a, b, c);
    }
    // store transposed: kvT[e][d]; M(=d) contiguous per lane -> packed b128
    const int e = e0 + (lane & 15);
    v8bf pk;
#pragma unroll
    for (int r = 0; r < 8; ++r) pk[r] = (bf16)c[r];
    *(v8bf*)(kvT + (bh * HDIM + e) * HDIM + d0 + ((lane >> 4) << 3)) = pk;
}

// ---------------------------------------------------------------------------
// 5) lin[b,n,h*64+e] = z[bh,n] * (qf[n,:] @ kv[:,e]) ; per-wave 16x64 strip.
// ---------------------------------------------------------------------------
__global__ void attn_out_kernel(const bf16* __restrict__ qf, const bf16* __restrict__ kvT,
                                const float* __restrict__ z, bf16* __restrict__ lin) {
    const int lane = threadIdx.x & 31, w = threadIdx.x >> 5;
    const int tile = blockIdx.x * 8 + w;         // 16 bh * 128 n-tiles
    const int bh = tile >> 7, n0 = (tile & 127) << 4;
    const bf16* A  = qf + (bh * NSEQ + n0) * HDIM;  // rows n, ld 64
    const bf16* Bt = kvT + bh * (HDIM * HDIM);      // rows e, ld 64
    v8f c[4] = {};
#pragma unroll
    for (int k0 = 0; k0 < HDIM; k0 += 32) {
        v16bf a = ldA(A, HDIM, 0, k0, lane);
#pragma unroll
        for (int t = 0; t < 4; ++t) {
            v16bf b = ldB(Bt, HDIM, t << 4, k0, lane);
            c[t] = wmma_bf16(a, b, c[t]);
        }
    }
    const int b_ = bh >> 2, h = bh & 3, col = lane & 15;
#pragma unroll
    for (int r = 0; r < 8; ++r) {
        int n = n0 + r + ((lane >> 4) << 3);
        float zz = z[bh * NSEQ + n];
        bf16* row = lin + (b_ * NSEQ + n) * HID + h * HDIM;
#pragma unroll
        for (int t = 0; t < 4; ++t) row[(t << 4) + col] = (bf16)(c[t][r] * zz);
    }
}

// ---------------------------------------------------------------------------
// 6) out = lin @ Wout^T + bout (f32) ; also writes the analytic reg loss.
// ---------------------------------------------------------------------------
__global__ void out_proj_kernel(const bf16* __restrict__ lin, const bf16* __restrict__ Wout,
                                const float* __restrict__ bout, float* __restrict__ out) {
    const int lane = threadIdx.x & 31, w = threadIdx.x >> 5;
    const int tile = blockIdx.x * 8 + w;         // 512 * 16 tiles
    const int g0 = (tile >> 4) << 4, j0 = (tile & 15) << 4;
    v8f c = {};
#pragma unroll
    for (int kk = 0; kk < HID; kk += 32) {
        __builtin_prefetch(lin + (g0 + (lane & 15)) * HID + kk + 64, 0, 3);
        v16bf a = ldA(lin, HID, g0, kk, lane);
        v16bf b = ldB(Wout, HID, j0, kk, lane);
        c = wmma_bf16(a, b, c);
    }
    const int j = j0 + (lane & 15);
    const float bj = bout[j];
#pragma unroll
    for (int r = 0; r < 8; ++r) {
        int g = g0 + r + ((lane >> 4) << 3);
        out[g * HID + j] = c[r] + bj;
    }
    if (blockIdx.x == 0 && threadIdx.x == 0) {
        // attn_reg_loss = REG_W * mean(|softmax(..)|) = REG_W / N (exact)
        out[NROWS * HID] = 1.0e-5f / (float)NSEQ;
    }
}

// ---------------------------------------------------------------------------
extern "C" void kernel_launch(void* const* d_in, const int* in_sizes, int n_in,
                              void* d_out, int out_size, void* d_ws, size_t ws_size,
                              hipStream_t stream) {
    (void)in_sizes; (void)n_in; (void)out_size; (void)ws_size;
    const float* x          = (const float*)d_in[0];
    const float* w_qkv_low  = (const float*)d_in[1];
    const float* b_qkv_low  = (const float*)d_in[2];
    const float* w_qkv_high = (const float*)d_in[3];
    const float* b_qkv_high = (const float*)d_in[4];
    const float* w_out_low  = (const float*)d_in[5];
    const float* b_out_low  = (const float*)d_in[6];
    const float* w_out_high = (const float*)d_in[7];
    const float* b_out_high = (const float*)d_in[8];
    // d_in[9] (adj_u), d_in[10] (adj_v) are dead: they only shift softmax
    // logits, and the loss depends only on softmax row sums (== 1).

    char* ws = (char*)d_ws;
    auto alloc = [&](size_t bytes) -> char* {
        char* p = ws; ws += (bytes + 255) & ~(size_t)255; return p;
    };
    bf16*  Weff = (bf16*)alloc((size_t)768 * HID * sizeof(bf16));   // 384 KB
    float* beff = (float*)alloc(768 * sizeof(float));
    bf16*  Wout = (bf16*)alloc((size_t)HID * HID * sizeof(bf16));   // 128 KB
    float* bout = (float*)alloc(HID * sizeof(float));
    bf16*  qf   = (bf16*)alloc((size_t)NBH * NSEQ * HDIM * sizeof(bf16)); // 4 MB
    bf16*  kfT  = (bf16*)alloc((size_t)NBH * HDIM * NSEQ * sizeof(bf16)); // 4 MB
    bf16*  vT   = (bf16*)alloc((size_t)NBH * HDIM * NSEQ * sizeof(bf16)); // 4 MB
    float* z    = (float*)alloc((size_t)NBH * NSEQ * sizeof(float));      // 128 KB
    bf16*  kvT  = (bf16*)alloc((size_t)NBH * HDIM * HDIM * sizeof(bf16)); // 128 KB
    bf16*  lin  = (bf16*)alloc((size_t)NROWS * HID * sizeof(bf16));       // 4 MB

    prep_qkv_w<<<768, 256, 0, stream>>>(w_qkv_low, b_qkv_low, w_qkv_high,
                                        b_qkv_high, Weff, beff);
    prep_out_w<<<256, 256, 0, stream>>>(w_out_low, b_out_low, w_out_high,
                                        b_out_high, Wout, bout);
    qkv_kernel<<<3072, 256, 0, stream>>>(x, Weff, beff, qf, kfT, vT);
    zsum_kernel<<<(NBH * NSEQ) / 256, 256, 0, stream>>>(kfT, z);
    kv_kernel<<<32, 256, 0, stream>>>(kfT, vT, kvT);
    attn_out_kernel<<<256, 256, 0, stream>>>(qf, kvT, z, lin);
    out_proj_kernel<<<1024, 256, 0, stream>>>(lin, Wout, bout, (float*)d_out);
}
